// FFTConv2d_88819923681993
// MI455X (gfx1250) — matrix operand order repaired and checked by reading, not compiled, not run on hardware
//
#include <hip/hip_runtime.h>

typedef __attribute__((ext_vector_type(16))) __bf16 v16bf;
typedef __attribute__((ext_vector_type(4)))  __bf16 v4bf;
typedef __attribute__((ext_vector_type(8)))  float  v8f;
typedef __attribute__((ext_vector_type(4), aligned(4))) float f4u;  // 4B-aligned float4

#define HWD   256
#define CIN   64
#define COUT  64
#define KS    31
#define KPAD  32
#define PADP  15
#define TILE  16
#define CCH   32          // channels staged per LDS chunk
#define SROWS 46          // TILE + KS - 1
#define SCOLS 48          // 46 padded to 48 for stride

// ---------------------------------------------------------------------------
// Pass 1: repack weights f32 [o][c][ki][kj] -> bf16 [c*31+ki][o][kj(32, pad 0)]
// ---------------------------------------------------------------------------
__global__ __launch_bounds__(256) void repack_weights(
    const float* __restrict__ w, __bf16* __restrict__ wbf) {
  int i = blockIdx.x * blockDim.x + threadIdx.x;
  const int N = CIN * KS * COUT * KPAD;
  if (i >= N) return;
  int kj = i & (KPAD - 1);
  int o  = (i >> 5) & (COUT - 1);
  int t  = i >> 11;                 // c*31 + ki
  int ki = t % KS;
  int c  = t / KS;
  float v = 0.f;
  if (kj < KS) v = w[(((size_t)o * CIN + c) * KS + ki) * KS + kj];
  wbf[i] = (__bf16)v;
}

// A fragment in contiguous k order (k = 16*(lane/16) + s), same as B (2x unaligned ds_load_b128)
__device__ __forceinline__ v16bf load_afrag(const __bf16* srow, int koffA) {
  v16bf a;
#pragma unroll
  for (int i = 0; i < 8; ++i) {
    a[i]     = srow[koffA + i];        // K 0..7  / 16..23
    a[i + 8] = srow[koffA + 8 + i];    // K 8..15 / 24..31
  }
  return a;
}

// ---------------------------------------------------------------------------
// Pass 2: implicit-GEMM direct conv, K-dim = kj taps (31 -> 32).
// Block: (b, 16x16 spatial tile), all 64 couts. 256 threads = 8 waves.
// Wave: cout group g = wv&3, row half = wv>>2 (8 rows), 8 accumulators.
// Sliding A window: afrag(ki+1, rt) == afrag(ki, rt+1), so per vertical tap
// only ONE new A fragment is loaded from LDS (8x DS-traffic reduction).
// ---------------------------------------------------------------------------
__global__ __launch_bounds__(256) void conv_wmma(
    const float* __restrict__ sigin, const __bf16* __restrict__ wbf,
    const float* __restrict__ bias,  float* __restrict__ out) {
  __shared__ __bf16 sig[CCH * SROWS * SCOLS];   // 141,312 B of the 320KB LDS

  const int tid   = threadIdx.x;
  const int lane  = tid & 31;
  const int wv    = tid >> 5;
  const int g     = wv & 3;                 // cout group (16 couts)
  const int rbase = (wv >> 2) * 8;          // first output row of this wave
  const int mcol  = lane & 15;              // M index (pixel) / N index (cout)
  const int koffA = (lane < 16) ? 0 : 16;   // A 16x32 bf16 lane layout (contiguous k, matches B)
  const int koffB = (lane < 16) ? 0 : 16;   // B 32x16 bf16 lane layout

  const int blk = blockIdx.x;
  const int w0  = (blk & 15) * TILE;
  const int h0  = ((blk >> 4) & 15) * TILE;
  const int b   = blk >> 8;

  // interior tiles never touch the zero-padding halo -> checkless fast path
  const bool interior = (h0 >= TILE) && (h0 <= HWD - 2 * TILE) &&
                        (w0 >= TILE) && (w0 <= HWD - 2 * TILE);

  const v8f vzero = {0.f, 0.f, 0.f, 0.f, 0.f, 0.f, 0.f, 0.f};
  v8f acc[8];
#pragma unroll
  for (int r = 0; r < 8; ++r) acc[r] = vzero;

#pragma unroll 1
  for (int cc = 0; cc < CIN; cc += CCH) {
    __syncthreads();
    // ---- stage signal halo patch (f32 -> bf16) into LDS ----
    if (interior) {
      const float* gbase =
          sigin + (((size_t)b * CIN + cc) * HWD + (h0 - PADP)) * HWD + (w0 - PADP);
      for (int idx = tid; idx < CCH * SROWS * (SCOLS / 4); idx += 256) {
        int q   = idx % (SCOLS / 4);
        int r   = idx / (SCOLS / 4);        // c*SROWS + row
        int c   = r / SROWS;
        int row = r - c * SROWS;
        f4u f = *(const f4u*)(gbase + (size_t)c * HWD * HWD + row * HWD + q * 4);
        v4bf hv = {(__bf16)f.x, (__bf16)f.y, (__bf16)f.z, (__bf16)f.w};
        *(v4bf*)(&sig[r * SCOLS + q * 4]) = hv;
      }
    } else {
      for (int idx = tid; idx < CCH * SROWS * SCOLS; idx += 256) {
        int c   = idx / (SROWS * SCOLS);
        int rem = idx - c * (SROWS * SCOLS);
        int row = rem / SCOLS;
        int col = rem - row * SCOLS;
        int h = h0 + row - PADP;
        int w = w0 + col - PADP;
        float v = 0.f;
        if ((unsigned)h < HWD && (unsigned)w < HWD)
          v = sigin[(((size_t)b * CIN + cc + c) * HWD + h) * HWD + w];
        sig[idx] = (__bf16)v;
      }
    }
    __syncthreads();

    // ---- K-loop: channels x vertical taps; WMMA over 32 kj taps ----
#pragma unroll 1
    for (int c = 0; c < CCH; ++c) {
      const __bf16* wbase =
          wbf + (((size_t)(cc + c) * KS) * COUT + g * 16 + mcol) * KPAD + koffB;
      __builtin_prefetch(wbase + (size_t)KS * COUT * KPAD, 0, 1);
      const __bf16* srowc = &sig[(c * SROWS + rbase) * SCOLS + mcol];

      // prime the 8-row sliding A window
      v16bf af[8];
#pragma unroll
      for (int rt = 0; rt < 8; ++rt)
        af[rt] = load_afrag(srowc + rt * SCOLS, koffA);

#pragma unroll   // full unroll: A-window rotation becomes pure SSA renaming
      for (int ki = 0; ki < KS; ++ki) {
        v16bf bfrag = *(const v16bf*)(wbase + (size_t)ki * COUT * KPAD);
#pragma unroll
        for (int rt = 0; rt < 8; ++rt)
          acc[rt] = __builtin_amdgcn_wmma_f32_16x16x32_bf16(
              false, af[rt], false, bfrag, (short)0, acc[rt], false, false);
        if (ki + 1 < KS) {        // slide window: one new row fragment only
#pragma unroll
          for (int rt = 0; rt < 7; ++rt) af[rt] = af[rt + 1];
          af[7] = load_afrag(srowc + (8 + ki) * SCOLS, koffA);
        }
      }
    }
  }

  // ---- epilogue: bias + store per C/D 16x16 f32 layout ----
  const int   n     = g * 16 + mcol;
  const float bv    = bias[n];
  const int   mbase = (lane < 16) ? 0 : 8;
#pragma unroll
  for (int rt = 0; rt < 8; ++rt) {
    int hrow = h0 + rbase + rt;
    float* orow =
        out + (((size_t)b * COUT + n) * HWD + hrow) * HWD + w0 + mbase;
#pragma unroll
    for (int v = 0; v < 8; ++v) orow[v] = acc[rt][v] + bv;
  }
}

// ---------------------------------------------------------------------------
extern "C" void kernel_launch(void* const* d_in, const int* in_sizes, int n_in,
                              void* d_out, int out_size, void* d_ws,
                              size_t ws_size, hipStream_t stream) {
  const float* signal = (const float*)d_in[0];  // [8,64,256,256] f32
  const float* weight = (const float*)d_in[1];  // [64,64,31,31]  f32
  const float* bias   = (const float*)d_in[2];  // [64]           f32
  float*  out = (float*)d_out;                  // [8,64,256,256] f32
  __bf16* wbf = (__bf16*)d_ws;                  // 7.75 MB repacked weights

  const int nrep = CIN * KS * COUT * KPAD;
  repack_weights<<<(nrep + 255) / 256, 256, 0, stream>>>(weight, wbf);

  const int nblk = 8 * (HWD / TILE) * (HWD / TILE);   // 2048
  conv_wmma<<<nblk, 256, 0, stream>>>(signal, wbf, bias, out);
}
